// MFFusionHead_61074434949487
// MI455X (gfx1250) — matrix operand order repaired and checked by reading, not compiled, not run on hardware
//
#include <hip/hip_runtime.h>
#include <hip/hip_bf16.h>

// ---------------------------------------------------------------------------
// MFFusionHead forward for gfx1250 (MI455X).  bf16 WMMA for all large GEMMs
// (implicit-GEMM 3x3 convs, MLPs, attention projections), f32 accumulate.
// ---------------------------------------------------------------------------

typedef __bf16 bf16_t;
typedef __attribute__((ext_vector_type(16))) __bf16 v16bf;
typedef __attribute__((ext_vector_type(8)))  float  v8f;

static constexpr int NB  = 2;            // batch
static constexpr int HD  = 180;          // H
static constexpr int WD  = 180;          // W
static constexpr int CI  = 384;          // in channels
static constexpr int CH  = 128;          // hidden channels
static constexpr int NCL = 10;           // classes
static constexpr int NP  = 200;          // proposals
static constexpr int NHD = 8;            // heads
static constexpr int HWS = HD * WD;      // 32400
static constexpr int MT  = NB * HWS;     // 64800 spatial rows

#define CDIV(a, b) (((a) + (b) - 1) / (b))

union FragAB { v16bf v; uint4 q[2]; };

__device__ __forceinline__ v8f wmma_bf16(const FragAB& a, const FragAB& b, v8f c) {
  return __builtin_amdgcn_wmma_f32_16x16x32_bf16(false, a.v, false, b.v, (short)0, c,
                                                 false, false);
}

// ---------------------------------------------------------------------------
// Packing / layout conversion
// ---------------------------------------------------------------------------

// x: [NB][CI][H][W] f32  ->  [NB][H][W][CI] bf16
__global__ void pack_x_nhwc(const float* __restrict__ x, bf16_t* __restrict__ o) {
  int idx = blockIdx.x * blockDim.x + threadIdx.x;
  if (idx >= MT * CI) return;
  int c = idx % CI;
  int r = idx / CI;              // b*HWS + pos
  int bb = r / HWS, pos = r % HWS;
  o[idx] = (bf16_t)x[((size_t)(bb * CI + c)) * HWS + pos];
}

// conv weight (Oreal, I, 3, 3) f32 -> [Opad][9*I] bf16, k = tap*I + i
__global__ void pack_convw(const float* __restrict__ w, bf16_t* __restrict__ o,
                           int Oreal, int Opad, int I) {
  int idx = blockIdx.x * blockDim.x + threadIdx.x;
  int Kt = 9 * I;
  if (idx >= Opad * Kt) return;
  int oo = idx / Kt, rem = idx % Kt;
  int tap = rem / I, i = rem % I;
  float v = (oo < Oreal) ? w[((size_t)(oo * I + i)) * 9 + tap] : 0.f;
  o[(size_t)oo * Kt + tap * I + i] = (bf16_t)v;
}

// w: [K][N] f32 -> [N][K] bf16 (transpose pack for WMMA B operand)
__global__ void pack_mat(const float* __restrict__ w, bf16_t* __restrict__ o, int K, int N) {
  int idx = blockIdx.x * blockDim.x + threadIdx.x;
  if (idx >= N * K) return;
  int n = idx / K, k = idx % K;
  o[idx] = (bf16_t)w[(size_t)k * N + n];
}

// ---------------------------------------------------------------------------
// Implicit-GEMM 3x3 SAME conv, NHWC bf16 in, WMMA bf16, f32 accumulate.
// Block: 256 threads = 8 waves; block tile 128M x (16*NT)N, wave tile 16 x 16*NT.
// ---------------------------------------------------------------------------
template <int NT>
__global__ __launch_bounds__(256) void conv3x3_wmma(
    const bf16_t* __restrict__ in, const bf16_t* __restrict__ wB,
    const float* __restrict__ bias, const float* __restrict__ gamma,
    const float* __restrict__ beta, float* __restrict__ outF,
    bf16_t* __restrict__ outH, int Cin, int Nst, int Nreal) {
  const int Ktot = 9 * Cin;
  const int tid = threadIdx.x;
  const int wave = tid >> 5, lane = tid & 31;
  const int g = lane >> 4, lm = lane & 15;
  const int mBase = (blockIdx.x * 8 + wave) * 16;
  const int nBase = blockIdx.y * (16 * NT);
  __shared__ bf16_t ldsB[NT * 16 * 32];

  int row = mBase + lm;
  bool vm = row < MT;
  int rc = vm ? row : 0;
  int bb = rc / HWS, rem = rc % HWS;
  int hh = rem / WD, xcol = rem % WD;

  v8f acc[NT] = {};

  for (int k0 = 0; k0 < Ktot; k0 += 32) {
    __syncthreads();
    for (int idx = tid; idx < NT * 16 * 4; idx += blockDim.x) {
      int rn = idx >> 2, kc = (idx & 3) * 8;
      *(uint4*)&ldsB[rn * 32 + kc] =
          *(const uint4*)&wB[(size_t)(nBase + rn) * Ktot + k0 + kc];
    }
    __syncthreads();

    int tap = k0 / Cin, cin0 = k0 - tap * Cin;
    int dy = tap / 3 - 1, dx = tap % 3 - 1;
    int yy = hh + dy, xx = xcol + dx;
    bool ok = vm && (unsigned)yy < (unsigned)HD && (unsigned)xx < (unsigned)WD;
    int yc = min(max(yy, 0), HD - 1), xc = min(max(xx, 0), WD - 1);
    const bf16_t* p = in + ((size_t)((bb * HD + yc) * WD + xc)) * Cin + cin0;
    FragAB a;
    if (ok) {
      a.q[0] = *(const uint4*)(p + 8 * g);
      a.q[1] = *(const uint4*)(p + 16 + 8 * g);
    } else {
      a.q[0] = uint4{0u, 0u, 0u, 0u};
      a.q[1] = uint4{0u, 0u, 0u, 0u};
    }
#pragma unroll
    for (int t = 0; t < NT; t++) {
      FragAB bfr;
      const bf16_t* bp = &ldsB[(t * 16 + lm) * 32 + 8 * g];
      bfr.q[0] = *(const uint4*)(bp);
      bfr.q[1] = *(const uint4*)(bp + 16);
      acc[t] = wmma_bf16(a, bfr, acc[t]);
    }
  }

#pragma unroll
  for (int t = 0; t < NT; t++) {
    int n = nBase + t * 16 + lm;
    bool vn = n < Nreal;
    float bi = vn ? bias[n] : 0.f;
    float ga = (gamma && vn) ? gamma[n] : 1.f;
    float be = (beta && vn) ? beta[n] : 0.f;
#pragma unroll
    for (int r = 0; r < 8; r++) {
      int mr = mBase + r + 8 * g;
      if (mr < MT && vn) {
        float v = acc[t][r] + bi;
        if (gamma) v = fmaxf(ga * v + be, 0.f);
        if (outF) outF[(size_t)mr * Nst + n] = v;
        if (outH) outH[(size_t)mr * Nst + n] = (bf16_t)v;
      }
    }
  }
}

// ---------------------------------------------------------------------------
// Generic GEMM: C[M][Nst] = relu?(A[M][K] @ B^T[N][K] + bias)
// ---------------------------------------------------------------------------
template <int NT>
__global__ __launch_bounds__(256) void gemm_wmma(
    const bf16_t* __restrict__ A, const bf16_t* __restrict__ wB,
    const float* __restrict__ bias, int relu, float* __restrict__ outF,
    bf16_t* __restrict__ outH, int M, int K, int Nst) {
  const int tid = threadIdx.x;
  const int wave = tid >> 5, lane = tid & 31;
  const int g = lane >> 4, lm = lane & 15;
  const int mBase = (blockIdx.x * 8 + wave) * 16;
  const int nBase = blockIdx.y * (16 * NT);
  __shared__ bf16_t ldsB[NT * 16 * 32];

  int row = mBase + lm;
  bool vm = row < M;

  v8f acc[NT] = {};

  for (int k0 = 0; k0 < K; k0 += 32) {
    __syncthreads();
    for (int idx = tid; idx < NT * 16 * 4; idx += blockDim.x) {
      int rn = idx >> 2, kc = (idx & 3) * 8;
      *(uint4*)&ldsB[rn * 32 + kc] =
          *(const uint4*)&wB[(size_t)(nBase + rn) * K + k0 + kc];
    }
    __syncthreads();

    FragAB a;
    if (vm) {
      const bf16_t* p = A + (size_t)row * K + k0;
      a.q[0] = *(const uint4*)(p + 8 * g);
      a.q[1] = *(const uint4*)(p + 16 + 8 * g);
      if (k0 + 32 < K) __builtin_prefetch((const void*)(p + 32), 0, 0);
    } else {
      a.q[0] = uint4{0u, 0u, 0u, 0u};
      a.q[1] = uint4{0u, 0u, 0u, 0u};
    }
#pragma unroll
    for (int t = 0; t < NT; t++) {
      FragAB bfr;
      const bf16_t* bp = &ldsB[(t * 16 + lm) * 32 + 8 * g];
      bfr.q[0] = *(const uint4*)(bp);
      bfr.q[1] = *(const uint4*)(bp + 16);
      acc[t] = wmma_bf16(a, bfr, acc[t]);
    }
  }

#pragma unroll
  for (int t = 0; t < NT; t++) {
    int n = nBase + t * 16 + lm;
    float bi = bias[n];
#pragma unroll
    for (int r = 0; r < 8; r++) {
      int mr = mBase + r + 8 * g;
      if (mr < M) {
        float v = acc[t][r] + bi;
        if (relu) v = fmaxf(v, 0.f);
        if (outF) outF[(size_t)mr * Nst + n] = v;
        if (outH) outH[(size_t)mr * Nst + n] = (bf16_t)v;
      }
    }
  }
}

// ---------------------------------------------------------------------------
// Heatmap sigmoid + 3x3 local-max NMS.  dense: [MT][16] f32 (only 10 valid).
// score: [NB][NCL][HWS] f32
// ---------------------------------------------------------------------------
__global__ void nms_score(const float* __restrict__ dense, float* __restrict__ score) {
  int idx = blockIdx.x * blockDim.x + threadIdx.x;
  if (idx >= MT) return;
  int bb = idx / HWS, pos = idx % HWS;
  int hh = pos / WD, xc = pos % WD;
  for (int k = 0; k < NCL; k++) {
    float c = dense[(size_t)idx * 16 + k];
    float mx = c;
    for (int dy = -1; dy <= 1; dy++)
      for (int dx = -1; dx <= 1; dx++) {
        int yy = hh + dy, xx = xc + dx;
        if ((unsigned)yy < (unsigned)HD && (unsigned)xx < (unsigned)WD)
          mx = fmaxf(mx, dense[((size_t)(bb * HWS) + yy * WD + xx) * 16 + k]);
      }
    float s = (c >= mx) ? 1.f / (1.f + __expf(-c)) : 0.f;
    score[(size_t)bb * NCL * HWS + (size_t)k * HWS + pos] = s;
  }
}

// Deterministic top-200 with lowest-index tie-break (matches lax.top_k order).
__global__ void topk200(const float* __restrict__ score, float* __restrict__ scratch,
                        int* __restrict__ top_pos) {
  const int N = NCL * HWS;
  int bb = blockIdx.x, tid = threadIdx.x;
  const float* s = score + (size_t)bb * N;
  float* sc = scratch + (size_t)bb * N;
  for (int i = tid; i < N; i += blockDim.x) sc[i] = s[i];
  __syncthreads();
  __shared__ float bv[1024];
  __shared__ int bi[1024];
  for (int p = 0; p < NP; p++) {
    float best = -3.0e38f;
    int bidx = N;
    for (int i = tid; i < N; i += blockDim.x) {
      float v = sc[i];
      if (v > best || (v == best && i < bidx)) { best = v; bidx = i; }
    }
    bv[tid] = best; bi[tid] = bidx;
    __syncthreads();
    for (int st = 512; st > 0; st >>= 1) {
      if (tid < st) {
        float ov = bv[tid + st]; int oi = bi[tid + st];
        if (ov > bv[tid] || (ov == bv[tid] && oi < bi[tid])) { bv[tid] = ov; bi[tid] = oi; }
      }
      __syncthreads();
    }
    if (tid == 0) {
      int w = bi[0];
      top_pos[bb * NP + p] = w % HWS;
      sc[w] = -3.0e38f;
    }
    __syncthreads();
  }
}

__global__ void gather_q(const float* __restrict__ feat, const int* __restrict__ top_pos,
                         float* __restrict__ q0f, bf16_t* __restrict__ q0h) {
  int idx = blockIdx.x * blockDim.x + threadIdx.x;
  if (idx >= NB * NP * CH) return;
  int c = idx % CH, bp = idx / CH, bb = bp / NP;
  int pos = top_pos[bp];
  float v = feat[((size_t)(bb * HWS + pos)) * CH + c];
  q0f[idx] = v;
  q0h[idx] = (bf16_t)v;
}

__global__ void gather_hm(const float* __restrict__ score, const int* __restrict__ top_pos,
                          float* __restrict__ qhm) {
  int idx = blockIdx.x * blockDim.x + threadIdx.x;
  if (idx >= NB * NCL * NP) return;
  int p = idx % NP, bk = idx / NP, bb = bk / NCL, k = bk % NCL;
  int pos = top_pos[bb * NP + p];
  qhm[idx] = score[(size_t)bb * NCL * HWS + (size_t)k * HWS + pos];
}

// pos2embed -> [rows][256] bf16.  idx_opt==null: row r is flat bev position r.
__global__ void posembed(const int* __restrict__ idx_opt, int rows, bf16_t* __restrict__ o) {
  int idx = blockIdx.x * blockDim.x + threadIdx.x;
  if (idx >= rows * 256) return;
  int e = idx % 256, r = idx / 256;
  int pos = idx_opt ? idx_opt[r] : r;
  int i = pos / WD, j = pos % WD;
  const float twopi = 6.283185307179586f;
  float coord = (e < 128) ? ((j + 0.5f) / (float)WD) : ((i + 0.5f) / (float)HD);
  int el = e & 127;
  float d = (float)(el >> 1) * (1.0f / 64.0f) + 1.0f;
  float ph = coord * twopi / d;
  float v = (el & 1) ? __cosf(ph) : __sinf(ph);
  o[(size_t)r * 256 + e] = (bf16_t)v;
}

// o = bf16(a + b[i % bmod])  (bmod==0 -> plain)
__global__ void add2_bf16(const float* __restrict__ a, const float* __restrict__ b,
                          bf16_t* __restrict__ o, int n, int bmod) {
  int i = blockIdx.x * blockDim.x + threadIdx.x;
  if (i >= n) return;
  int bi = bmod ? (i % bmod) : i;
  o[i] = (bf16_t)(a[i] + b[bi]);
}

// Self-attention, L=200 keys, d=16/head, online softmax per query thread.
__global__ void attn_self(const float* __restrict__ Q, const float* __restrict__ Kp,
                          const float* __restrict__ Vp, float* __restrict__ Of,
                          bf16_t* __restrict__ Oh) {
  int bh = blockIdx.x, bb = bh / NHD, h = bh % NHD;
  int qi = threadIdx.x;
  if (qi >= NP) return;
  const float* qp = Q + ((size_t)(bb * NP + qi)) * CH + h * 16;
  float qv[16], ac[16];
  for (int d = 0; d < 16; d++) { qv[d] = qp[d]; ac[d] = 0.f; }
  float m = -3.0e38f, l = 0.f;
  for (int k = 0; k < NP; k++) {
    const float* kp = Kp + ((size_t)(bb * NP + k)) * CH + h * 16;
    float s = 0.f;
    for (int d = 0; d < 16; d++) s += qv[d] * kp[d];
    s *= 0.25f;
    float mn = fmaxf(m, s);
    float corr = __expf(m - mn), w = __expf(s - mn);
    l = l * corr + w;
    const float* vp = Vp + ((size_t)(bb * NP + k)) * CH + h * 16;
    for (int d = 0; d < 16; d++) ac[d] = ac[d] * corr + w * vp[d];
    m = mn;
  }
  float inv = 1.f / l;
  size_t ob = ((size_t)(bb * NP + qi)) * CH + h * 16;
  for (int d = 0; d < 16; d++) {
    float v = ac[d] * inv;
    Of[ob + d] = v;
    Oh[ob + d] = (bf16_t)v;
  }
}

// Cross-attention to the 32400-position BEV memory (K/V in bf16).
__global__ void attn_cross(const float* __restrict__ Q, const bf16_t* __restrict__ Kp,
                           const bf16_t* __restrict__ Vp, float* __restrict__ Of,
                           bf16_t* __restrict__ Oh) {
  int bh = blockIdx.x, bb = bh / NHD, h = bh % NHD;
  int qi = threadIdx.x;
  if (qi >= NP) return;
  const float* qp = Q + ((size_t)(bb * NP + qi)) * CH + h * 16;
  float qv[16], ac[16];
  for (int d = 0; d < 16; d++) { qv[d] = qp[d]; ac[d] = 0.f; }
  float m = -3.0e38f, l = 0.f;
  for (int k = 0; k < HWS; k++) {
    const bf16_t* kp = Kp + ((size_t)(bb * HWS + k)) * CH + h * 16;
    float s = 0.f;
    for (int d = 0; d < 16; d++) s += qv[d] * (float)kp[d];
    s *= 0.25f;
    float mn = fmaxf(m, s);
    float corr = __expf(m - mn), w = __expf(s - mn);
    l = l * corr + w;
    const bf16_t* vp = Vp + ((size_t)(bb * HWS + k)) * CH + h * 16;
    for (int d = 0; d < 16; d++) ac[d] = ac[d] * corr + w * (float)vp[d];
    m = mn;
  }
  float inv = 1.f / l;
  size_t ob = ((size_t)(bb * NP + qi)) * CH + h * 16;
  for (int d = 0; d < 16; d++) {
    float v = ac[d] * inv;
    Of[ob + d] = v;
    Oh[ob + d] = (bf16_t)v;
  }
}

// y = LN(a + b) * g + be, per 128-wide row.  grid = rows, block = 128.
__global__ void ln_res(const float* __restrict__ A, const float* __restrict__ Bt,
                       const float* __restrict__ g, const float* __restrict__ be,
                       float* __restrict__ outF, bf16_t* __restrict__ outH) {
  int r = blockIdx.x, c = threadIdx.x;
  float x = A[(size_t)r * CH + c] + Bt[(size_t)r * CH + c];
  __shared__ float s1[CH], s2[CH];
  s1[c] = x;
  s2[c] = x * x;
  __syncthreads();
  for (int st = 64; st > 0; st >>= 1) {
    if (c < st) { s1[c] += s1[c + st]; s2[c] += s2[c + st]; }
    __syncthreads();
  }
  float mean = s1[0] * (1.0f / CH);
  float var = s2[0] * (1.0f / CH) - mean * mean;
  float y = g[c] * (x - mean) * rsqrtf(var + 1e-5f) + be[c];
  outF[(size_t)r * CH + c] = y;
  if (outH) outH[(size_t)r * CH + c] = (bf16_t)y;
}

// SeparateHead conv1d(k=3,SAME) stage 1: relu(g*(conv(q3)+b)+beta), all 6 heads.
__global__ void head_stage1(const float* __restrict__ q3, const float* __restrict__ w1,
                            const float* __restrict__ b1, const float* __restrict__ g1,
                            const float* __restrict__ be1, float* __restrict__ hid) {
  int idx = blockIdx.x * blockDim.x + threadIdx.x;  // ((i*NB+b)*CH+co)*NP+p
  if (idx >= 6 * NB * CH * NP) return;
  int p = idx % NP;
  int co = (idx / NP) % CH;
  int bb = (idx / (NP * CH)) % NB;
  int i = idx / (NP * CH * NB);
  float acc = 0.f;
  const float* wp = w1 + ((size_t)(i * CH + co)) * CH * 3;
  for (int ci = 0; ci < CH; ci++) {
    for (int t = 0; t < 3; t++) {
      int pp2 = p + t - 1;
      if ((unsigned)pp2 < (unsigned)NP)
        acc += wp[ci * 3 + t] * q3[((size_t)(bb * NP + pp2)) * CH + ci];
    }
  }
  acc += b1[i * CH + co];
  hid[idx] = fmaxf(g1[i * CH + co] * acc + be1[i * CH + co], 0.f);
}

// Stage 2: per-head conv1d to output dims, fuse heatmap scores into class logits.
__global__ void head_stage2(const float* __restrict__ hid, const float* __restrict__ w2,
                            const float* __restrict__ b2, const float* __restrict__ qhm,
                            float* __restrict__ out) {
  int idx = blockIdx.x * blockDim.x + threadIdx.x;  // (b*20+och)*NP+p
  if (idx >= NB * 20 * NP) return;
  int p = idx % NP;
  int och = (idx / NP) % 20;
  int bb = idx / (NP * 20);
  const int offs[7] = {0, 2, 3, 6, 8, 10, 20};
  int i = 0;
  while (och >= offs[i + 1]) i++;
  int c = och - offs[i];
  float acc = b2[i * NCL + c];
  const float* wp = w2 + ((size_t)(i * NCL + c)) * CH * 3;
  const float* hb = hid + ((size_t)((i * NB + bb) * CH)) * NP;
  for (int ci = 0; ci < CH; ci++) {
    for (int t = 0; t < 3; t++) {
      int pp2 = p + t - 1;
      if ((unsigned)pp2 < (unsigned)NP) acc += wp[ci * 3 + t] * hb[(size_t)ci * NP + pp2];
    }
  }
  if (i == 5) acc += qhm[((size_t)(bb * NCL + c)) * NP + p];
  out[idx] = acc;
}

// ---------------------------------------------------------------------------
// Host orchestration
// ---------------------------------------------------------------------------
extern "C" void kernel_launch(void* const* d_in, const int* in_sizes, int n_in,
                              void* d_out, int out_size, void* d_ws, size_t ws_size,
                              hipStream_t stream) {
  (void)in_sizes; (void)n_in; (void)out_size; (void)ws_size;
  const float* x        = (const float*)d_in[0];
  const float* w_shared = (const float*)d_in[1];
  const float* b_shared = (const float*)d_in[2];
  const float* w_hm1    = (const float*)d_in[3];
  const float* b_hm1    = (const float*)d_in[4];
  const float* g_hm1    = (const float*)d_in[5];
  const float* be_hm1   = (const float*)d_in[6];
  const float* w_hm2    = (const float*)d_in[7];
  const float* b_hm2    = (const float*)d_in[8];
  // d_in[9..14]: frontmap head — computed then deleted in reference, unused.
  const float* w_be1    = (const float*)d_in[15];
  const float* b_be1    = (const float*)d_in[16];
  const float* w_be2    = (const float*)d_in[17];
  const float* b_be2    = (const float*)d_in[18];
  const float* w_attn_s = (const float*)d_in[19];
  const float* b_attn_s = (const float*)d_in[20];
  const float* w_out_s  = (const float*)d_in[21];
  const float* b_out_s  = (const float*)d_in[22];
  const float* w_attn_c = (const float*)d_in[23];
  const float* b_attn_c = (const float*)d_in[24];
  const float* w_out_c  = (const float*)d_in[25];
  const float* b_out_c  = (const float*)d_in[26];
  const float* ln1_g    = (const float*)d_in[27];
  const float* ln1_b    = (const float*)d_in[28];
  const float* ln2_g    = (const float*)d_in[29];
  const float* ln2_b    = (const float*)d_in[30];
  const float* ln3_g    = (const float*)d_in[31];
  const float* ln3_b    = (const float*)d_in[32];
  const float* w_ff1    = (const float*)d_in[33];
  const float* b_ff1    = (const float*)d_in[34];
  const float* w_ff2    = (const float*)d_in[35];
  const float* b_ff2    = (const float*)d_in[36];
  const float* w_head1  = (const float*)d_in[37];
  const float* b_head1  = (const float*)d_in[38];
  const float* g_head1  = (const float*)d_in[39];
  const float* be_head1 = (const float*)d_in[40];
  const float* w_head2  = (const float*)d_in[41];
  const float* b_head2  = (const float*)d_in[42];
  float* out = (float*)d_out;

  char* base = (char*)d_ws;
  size_t off = 0;
  auto alloc = [&](size_t bytes) -> void* {
    void* p = base + off;
    off = (off + bytes + 255) & ~(size_t)255;
    return p;
  };

  bf16_t* xh    = (bf16_t*)alloc((size_t)MT * CI * 2);
  bf16_t* wbsh  = (bf16_t*)alloc((size_t)CH * 9 * CI * 2);
  bf16_t* wbh1  = (bf16_t*)alloc((size_t)CH * 9 * CH * 2);
  bf16_t* wbh2  = (bf16_t*)alloc((size_t)16 * 9 * CH * 2);
  bf16_t* wbe1  = (bf16_t*)alloc((size_t)CH * 256 * 2);
  bf16_t* wbe2  = (bf16_t*)alloc((size_t)CH * CH * 2);
  bf16_t* ws0   = (bf16_t*)alloc((size_t)CH * CH * 2);
  bf16_t* ws1   = (bf16_t*)alloc((size_t)CH * CH * 2);
  bf16_t* ws2   = (bf16_t*)alloc((size_t)CH * CH * 2);
  bf16_t* wos   = (bf16_t*)alloc((size_t)CH * CH * 2);
  bf16_t* wc0   = (bf16_t*)alloc((size_t)CH * CH * 2);
  bf16_t* wc1   = (bf16_t*)alloc((size_t)CH * CH * 2);
  bf16_t* wc2   = (bf16_t*)alloc((size_t)CH * CH * 2);
  bf16_t* woc   = (bf16_t*)alloc((size_t)CH * CH * 2);
  bf16_t* wf1   = (bf16_t*)alloc((size_t)256 * CH * 2);
  bf16_t* wf2   = (bf16_t*)alloc((size_t)CH * 256 * 2);
  float*  featf = (float*)alloc((size_t)MT * CH * 4);
  bf16_t* feath = (bf16_t*)alloc((size_t)MT * CH * 2);
  bf16_t* hidh  = (bf16_t*)alloc((size_t)MT * CH * 2);
  float*  dense = (float*)alloc((size_t)MT * 16 * 4);
  float*  score = (float*)alloc((size_t)NB * NCL * HWS * 4);
  float*  scr   = (float*)alloc((size_t)NB * NCL * HWS * 4);
  int*    tpos  = (int*)alloc((size_t)NB * NP * 4);
  float*  qhm   = (float*)alloc((size_t)NB * NCL * NP * 4);
  bf16_t* pe    = (bf16_t*)alloc((size_t)HWS * 256 * 2);
  bf16_t* t1h   = (bf16_t*)alloc((size_t)HWS * CH * 2);
  float*  kpef  = (float*)alloc((size_t)HWS * CH * 4);
  bf16_t* peq   = (bf16_t*)alloc((size_t)NB * NP * 256 * 2);
  bf16_t* tq1h  = (bf16_t*)alloc((size_t)NB * NP * CH * 2);
  float*  qpef  = (float*)alloc((size_t)NB * NP * CH * 4);
  float*  q0f   = (float*)alloc((size_t)NB * NP * CH * 4);
  bf16_t* q0h   = (bf16_t*)alloc((size_t)NB * NP * CH * 2);
  bf16_t* qqpeh = (bf16_t*)alloc((size_t)NB * NP * CH * 2);
  float*  qsf   = (float*)alloc((size_t)NB * NP * CH * 4);
  float*  ksf   = (float*)alloc((size_t)NB * NP * CH * 4);
  float*  vsf   = (float*)alloc((size_t)NB * NP * CH * 4);
  float*  aof   = (float*)alloc((size_t)NB * NP * CH * 4);
  bf16_t* aoh   = (bf16_t*)alloc((size_t)NB * NP * CH * 2);
  float*  sof   = (float*)alloc((size_t)NB * NP * CH * 4);
  float*  q1f   = (float*)alloc((size_t)NB * NP * CH * 4);
  bf16_t* qq1ph = (bf16_t*)alloc((size_t)NB * NP * CH * 2);
  float*  qcqf  = (float*)alloc((size_t)NB * NP * CH * 4);
  bf16_t* mkph  = (bf16_t*)alloc((size_t)MT * CH * 2);
  bf16_t* kch   = (bf16_t*)alloc((size_t)MT * CH * 2);
  bf16_t* vch   = (bf16_t*)alloc((size_t)MT * CH * 2);
  float*  cof   = (float*)alloc((size_t)NB * NP * CH * 4);
  bf16_t* coh   = (bf16_t*)alloc((size_t)NB * NP * CH * 2);
  float*  cxf   = (float*)alloc((size_t)NB * NP * CH * 4);
  float*  q2f   = (float*)alloc((size_t)NB * NP * CH * 4);
  bf16_t* q2h   = (bf16_t*)alloc((size_t)NB * NP * CH * 2);
  bf16_t* ff1h  = (bf16_t*)alloc((size_t)NB * NP * 256 * 2);
  float*  ffnf  = (float*)alloc((size_t)NB * NP * CH * 4);
  float*  q3f   = (float*)alloc((size_t)NB * NP * CH * 4);
  float*  hid6  = (float*)alloc((size_t)6 * NB * CH * NP * 4);

  const int M4 = NB * NP;  // 400 query rows

  // ---- packing ------------------------------------------------------------
  pack_x_nhwc<<<CDIV(MT * CI, 256), 256, 0, stream>>>(x, xh);
  pack_convw<<<CDIV(CH * 9 * CI, 256), 256, 0, stream>>>(w_shared, wbsh, CH, CH, CI);
  pack_convw<<<CDIV(CH * 9 * CH, 256), 256, 0, stream>>>(w_hm1, wbh1, CH, CH, CH);
  pack_convw<<<CDIV(16 * 9 * CH, 256), 256, 0, stream>>>(w_hm2, wbh2, NCL, 16, CH);
  pack_mat<<<CDIV(CH * 256, 256), 256, 0, stream>>>(w_be1, wbe1, 256, CH);
  pack_mat<<<CDIV(CH * CH, 256), 256, 0, stream>>>(w_be2, wbe2, CH, CH);
  pack_mat<<<CDIV(CH * CH, 256), 256, 0, stream>>>(w_attn_s + 0 * CH * CH, ws0, CH, CH);
  pack_mat<<<CDIV(CH * CH, 256), 256, 0, stream>>>(w_attn_s + 1 * CH * CH, ws1, CH, CH);
  pack_mat<<<CDIV(CH * CH, 256), 256, 0, stream>>>(w_attn_s + 2 * CH * CH, ws2, CH, CH);
  pack_mat<<<CDIV(CH * CH, 256), 256, 0, stream>>>(w_out_s, wos, CH, CH);
  pack_mat<<<CDIV(CH * CH, 256), 256, 0, stream>>>(w_attn_c + 0 * CH * CH, wc0, CH, CH);
  pack_mat<<<CDIV(CH * CH, 256), 256, 0, stream>>>(w_attn_c + 1 * CH * CH, wc1, CH, CH);
  pack_mat<<<CDIV(CH * CH, 256), 256, 0, stream>>>(w_attn_c + 2 * CH * CH, wc2, CH, CH);
  pack_mat<<<CDIV(CH * CH, 256), 256, 0, stream>>>(w_out_c, woc, CH, CH);
  pack_mat<<<CDIV(256 * CH, 256), 256, 0, stream>>>(w_ff1, wf1, CH, 256);
  pack_mat<<<CDIV(CH * 256, 256), 256, 0, stream>>>(w_ff2, wf2, 256, CH);

  // ---- convs (implicit GEMM, WMMA) ---------------------------------------
  dim3 cg(CDIV(MT, 128), 2);
  conv3x3_wmma<4><<<cg, 256, 0, stream>>>(xh, wbsh, b_shared, nullptr, nullptr,
                                          featf, feath, CI, CH, CH);
  conv3x3_wmma<4><<<cg, 256, 0, stream>>>(feath, wbh1, b_hm1, g_hm1, be_hm1,
                                          nullptr, hidh, CH, CH, CH);
  conv3x3_wmma<1><<<dim3(CDIV(MT, 128), 1), 256, 0, stream>>>(
      hidh, wbh2, b_hm2, nullptr, nullptr, dense, nullptr, CH, 16, NCL);

  // ---- NMS + top-k + gathers ---------------------------------------------
  nms_score<<<CDIV(MT, 256), 256, 0, stream>>>(dense, score);
  topk200<<<NB, 1024, 0, stream>>>(score, scr, tpos);
  gather_q<<<CDIV(M4 * CH, 256), 256, 0, stream>>>(featf, tpos, q0f, q0h);
  gather_hm<<<CDIV(NB * NCL * NP, 256), 256, 0, stream>>>(score, tpos, qhm);

  // ---- positional embeddings (pos2embed + MLP) ---------------------------
  posembed<<<CDIV(HWS * 256, 256), 256, 0, stream>>>(nullptr, HWS, pe);
  posembed<<<CDIV(M4 * 256, 256), 256, 0, stream>>>(tpos, M4, peq);
  gemm_wmma<4><<<dim3(CDIV(HWS, 128), 2), 256, 0, stream>>>(pe, wbe1, b_be1, 1,
                                                            nullptr, t1h, HWS, 256, CH);
  gemm_wmma<4><<<dim3(CDIV(HWS, 128), 2), 256, 0, stream>>>(t1h, wbe2, b_be2, 0,
                                                            kpef, nullptr, HWS, CH, CH);
  gemm_wmma<4><<<dim3(CDIV(M4, 128), 2), 256, 0, stream>>>(peq, wbe1, b_be1, 1,
                                                           nullptr, tq1h, M4, 256, CH);
  gemm_wmma<4><<<dim3(CDIV(M4, 128), 2), 256, 0, stream>>>(tq1h, wbe2, b_be2, 0,
                                                           qpef, nullptr, M4, CH, CH);

  // ---- decoder: self-attention -------------------------------------------
  add2_bf16<<<CDIV(M4 * CH, 256), 256, 0, stream>>>(q0f, qpef, qqpeh, M4 * CH, 0);
  gemm_wmma<4><<<dim3(CDIV(M4, 128), 2), 256, 0, stream>>>(qqpeh, ws0, b_attn_s + 0 * CH,
                                                           0, qsf, nullptr, M4, CH, CH);
  gemm_wmma<4><<<dim3(CDIV(M4, 128), 2), 256, 0, stream>>>(qqpeh, ws1, b_attn_s + 1 * CH,
                                                           0, ksf, nullptr, M4, CH, CH);
  gemm_wmma<4><<<dim3(CDIV(M4, 128), 2), 256, 0, stream>>>(q0h, ws2, b_attn_s + 2 * CH,
                                                           0, vsf, nullptr, M4, CH, CH);
  attn_self<<<NB * NHD, 256, 0, stream>>>(qsf, ksf, vsf, aof, aoh);
  gemm_wmma<4><<<dim3(CDIV(M4, 128), 2), 256, 0, stream>>>(aoh, wos, b_out_s, 0,
                                                           sof, nullptr, M4, CH, CH);
  ln_res<<<M4, CH, 0, stream>>>(q0f, sof, ln1_g, ln1_b, q1f, nullptr);

  // ---- decoder: cross-attention to BEV memory ----------------------------
  add2_bf16<<<CDIV(M4 * CH, 256), 256, 0, stream>>>(q1f, qpef, qq1ph, M4 * CH, 0);
  gemm_wmma<4><<<dim3(CDIV(M4, 128), 2), 256, 0, stream>>>(qq1ph, wc0, b_attn_c + 0 * CH,
                                                           0, qcqf, nullptr, M4, CH, CH);
  add2_bf16<<<CDIV(MT * CH, 256), 256, 0, stream>>>(featf, kpef, mkph, MT * CH, HWS * CH);
  gemm_wmma<4><<<dim3(CDIV(MT, 128), 2), 256, 0, stream>>>(mkph, wc1, b_attn_c + 1 * CH,
                                                           0, nullptr, kch, MT, CH, CH);
  gemm_wmma<4><<<dim3(CDIV(MT, 128), 2), 256, 0, stream>>>(feath, wc2, b_attn_c + 2 * CH,
                                                           0, nullptr, vch, MT, CH, CH);
  attn_cross<<<NB * NHD, 256, 0, stream>>>(qcqf, kch, vch, cof, coh);
  gemm_wmma<4><<<dim3(CDIV(M4, 128), 2), 256, 0, stream>>>(coh, woc, b_out_c, 0,
                                                           cxf, nullptr, M4, CH, CH);
  ln_res<<<M4, CH, 0, stream>>>(q1f, cxf, ln2_g, ln2_b, q2f, q2h);

  // ---- decoder: FFN -------------------------------------------------------
  gemm_wmma<4><<<dim3(CDIV(M4, 128), 4), 256, 0, stream>>>(q2h, wf1, b_ff1, 1,
                                                           nullptr, ff1h, M4, CH, 256);
  gemm_wmma<4><<<dim3(CDIV(M4, 128), 2), 256, 0, stream>>>(ff1h, wf2, b_ff2, 0,
                                                           ffnf, nullptr, M4, 256, CH);
  ln_res<<<M4, CH, 0, stream>>>(q2f, ffnf, ln3_g, ln3_b, q3f, nullptr);

  // ---- SeparateHead conv1d heads -----------------------------------------
  head_stage1<<<CDIV(6 * NB * CH * NP, 256), 256, 0, stream>>>(q3f, w_head1, b_head1,
                                                               g_head1, be_head1, hid6);
  head_stage2<<<CDIV(NB * 20 * NP, 256), 256, 0, stream>>>(hid6, w_head2, b_head2,
                                                           qhm, out);
}
